// VectorQuantizer_88742614270487
// MI455X (gfx1250) — matrix operand order repaired and checked by reading, not compiled, not run on hardware
//
#include <hip/hip_runtime.h>
#include <hip/hip_bf16.h>
#include <math.h>

// ---------------- problem constants ----------------
#define N_ROWS 32768      // B*T = 16*2048
#define DIM    64
#define KCODES 8192
#define TILE_K 256                    // codes staged per TDM tile (32 KB bf16)
#define KTILES (KCODES / TILE_K)      // 32
#define BETA_F 0.01f

// ---------------- vector types (trivial, union-safe) ----------------
typedef __bf16 bf16_t;
typedef bf16_t v16bf __attribute__((ext_vector_type(16)));
typedef float  v8f   __attribute__((ext_vector_type(8)));
typedef float  f2    __attribute__((ext_vector_type(2)));
typedef unsigned int u4    __attribute__((ext_vector_type(4)));
typedef unsigned int u32x4 __attribute__((ext_vector_type(4)));
typedef int          i32x8 __attribute__((ext_vector_type(8)));
typedef int          i32x4 __attribute__((ext_vector_type(4)));

union V16 { v16bf v; u4 u[2]; };
union PK2 { bf16_t h[2]; unsigned int u; };

__device__ __forceinline__ float wave_sum(float v) {
  #pragma unroll
  for (int m = 16; m >= 1; m >>= 1) v += __shfl_xor(v, m, 32);
  return v;
}

// ---------------- TDM: 2D tile (TILE_K rows x 64 elems, 2B) global -> LDS ----
__device__ __forceinline__ void tdm_issue(const void* gaddr, unsigned int ldsOff) {
  unsigned long long ga = (unsigned long long)gaddr;
  u32x4 g0;
  g0[0] = 1u;                                   // count=1, user desc, no gather
  g0[1] = ldsOff;                               // lds_addr [63:32]
  g0[2] = (unsigned int)ga;                     // global_addr low
  g0[3] = (unsigned int)(ga >> 32) | (2u << 30);// global_addr high | type=2
  i32x8 g1;
  g1[0] = (int)(1u << 16);                      // data_size=1 -> 2 bytes/elem
  g1[1] = (int)(((unsigned)DIM) << 16);         // tensor_dim0[15:0]=64
  g1[2] = (int)(((unsigned)KCODES & 0xFFFFu) << 16); // tensor_dim1[15:0]=8192
  g1[3] = (int)(((unsigned)DIM) << 16);         // tile_dim0=64
  g1[4] = (int)TILE_K;                          // tile_dim1=256
  g1[5] = (int)DIM;                             // tensor_dim0_stride=64
  g1[6] = 0;
  g1[7] = 0;
  i32x4 gz4 = {0, 0, 0, 0};
  i32x8 gz8 = {0, 0, 0, 0, 0, 0, 0, 0};
  // amdgpu-toolchain (clang-23) 6-arg form
  __builtin_amdgcn_tensor_load_to_lds(g0, g1, gz4, gz4, gz8, 0);
}

// ---------------- kernel 1: normalize z rows, emit bf16 -------------------
__global__ __launch_bounds__(256) void vq_nrm_z(const float* __restrict__ z,
                                                bf16_t* __restrict__ zb) {
  const int wid = threadIdx.x >> 5, lane = threadIdx.x & 31;
  const int row = blockIdx.x * 8 + wid;
  f2 v = *(const f2*)(z + row * DIM + lane * 2);
  float ss = wave_sum(v.x * v.x + v.y * v.y);
  float r = rsqrtf(ss);
  PK2 p; p.h[0] = (bf16_t)(v.x * r); p.h[1] = (bf16_t)(v.y * r);
  ((unsigned int*)zb)[row * 32 + lane] = p.u;
}

// ---------------- kernel 2: normalize codebook; fp32 + bf16 ----------------
__global__ __launch_bounds__(256) void vq_nrm_e(const float* __restrict__ e,
                                                float* __restrict__ en,
                                                bf16_t* __restrict__ eb) {
  const int wid = threadIdx.x >> 5, lane = threadIdx.x & 31;
  const int row = blockIdx.x * 8 + wid;
  f2 v = *(const f2*)(e + row * DIM + lane * 2);
  float ss = wave_sum(v.x * v.x + v.y * v.y);
  float r = rsqrtf(ss);
  f2 o; o.x = v.x * r; o.y = v.y * r;
  *(f2*)(en + row * DIM + lane * 2) = o;
  PK2 p; p.h[0] = (bf16_t)o.x; p.h[1] = (bf16_t)o.y;
  ((unsigned int*)eb)[row * 32 + lane] = p.u;
}

// ---------------- kernel 3: WMMA argmax over the codebook -------------------
// 4 waves/block; each wave owns 2 row-tiles (32 rows); block = 128 rows ->
// 256 workgroups. Codebook tiles (256 codes x 64 d, bf16) double-buffered in
// LDS via the Tensor Data Mover. Both operands are unit-norm, so
// argmin dist == argmax dot (the ||e||^2 residual is O(ulp), far below the
// bf16 quantization noise of the dot itself).
__global__ __launch_bounds__(128) void vq_argmax(const bf16_t* __restrict__ zb,
                                                 const bf16_t* __restrict__ eb,
                                                 int* __restrict__ bestIdx) {
  __shared__ __align__(16) unsigned char tiles[2][TILE_K * DIM * 2]; // 2x32KB
  const int wid = threadIdx.x >> 5, lane = threadIdx.x & 31;
  const int waveRow = blockIdx.x * 128 + wid * 32;
  const int hi = (lane >> 4) & 1;   // 0: lanes 0-15, 1: lanes 16-31
  const int ln = lane & 15;

  // Resident A operands: [row-tile r][k-half h], in contiguous k order, same as B:
// lanes 0-15 hold d {0..15}, lanes 16-31 hold d {16..31} of 32h+.
  V16 A[2][2];
  const unsigned char* zbytes = (const unsigned char*)zb;
  #pragma unroll
  for (int r = 0; r < 2; ++r) {
    const int row = waveRow + r * 16 + ln;
    #pragma unroll
    for (int h = 0; h < 2; ++h) {
      const u4* p = (const u4*)(zbytes + row * (DIM * 2) + h * 64 + hi * 32);
      A[r][h].u[0] = p[0];
      A[r][h].u[1] = p[1];          // +16 bytes (d+8)
    }
  }

  float best[2][8]; int bidx[2][8];
  #pragma unroll
  for (int r = 0; r < 2; ++r)
    #pragma unroll
    for (int j = 0; j < 8; ++j) { best[r][j] = -3.402823e38f; bidx[r][j] = 0; }

  const unsigned char* ebytes = (const unsigned char*)eb;
  const unsigned int ldsBase = (unsigned int)(unsigned long long)(&tiles[0][0]);

  if (wid == 0) tdm_issue(ebytes, ldsBase);            // prologue: tile 0
  __builtin_amdgcn_s_wait_tensorcnt(0);
  __syncthreads();

  for (int t = 0; t < KTILES; ++t) {
    if (wid == 0 && (t + 1) < KTILES)
      tdm_issue(ebytes + (size_t)(t + 1) * TILE_K * DIM * 2,
                ldsBase + (unsigned)(((t + 1) & 1) * TILE_K * DIM * 2));

    const unsigned char* buf = tiles[t & 1];
    const int kBase = t * TILE_K;
    #pragma unroll 2
    for (int ct = 0; ct < TILE_K / 16; ++ct) {
      const int code = kBase + ct * 16 + ln;
      // B (32x16 bf16): lane=column; lanes 0-15 hold contraction d 0..15,
      // lanes 16-31 hold d 16..31 -> 32 contiguous bytes of the code's row.
      V16 B0, B1;
      const unsigned char* rowp = buf + (ct * 16 + ln) * (DIM * 2) + hi * 32;
      const u4* p0 = (const u4*)(rowp);          // d-half 0
      const u4* p1 = (const u4*)(rowp + 64);     // d-half 1
      B0.u[0] = p0[0]; B0.u[1] = p0[1];
      B1.u[0] = p1[0]; B1.u[1] = p1[1];
      #pragma unroll
      for (int r = 0; r < 2; ++r) {
        v8f c = {};
        c = __builtin_amdgcn_wmma_f32_16x16x32_bf16(false, A[r][0].v, false, B0.v,
                                                    (short)0, c, false, false);
        c = __builtin_amdgcn_wmma_f32_16x16x32_bf16(false, A[r][1].v, false, B1.v,
                                                    (short)0, c, false, false);
        #pragma unroll
        for (int j = 0; j < 8; ++j) {
          if (c[j] > best[r][j]) { best[r][j] = c[j]; bidx[r][j] = code; }
        }
      }
    }
    __syncthreads();                              // all done reading buf[t&1]
    __builtin_amdgcn_s_wait_tensorcnt(0);         // wave0: tile t+1 landed
    __syncthreads();
  }

  // Cross-lane argmax within each 16-lane half (same row, different columns).
  #pragma unroll
  for (int r = 0; r < 2; ++r) {
    #pragma unroll
    for (int j = 0; j < 8; ++j) {
      float b = best[r][j]; int i = bidx[r][j];
      #pragma unroll
      for (int m = 8; m >= 1; m >>= 1) {
        const float ob = __shfl_xor(b, m, 32);
        const int   oi = __shfl_xor(i, m, 32);
        if (ob > b || (ob == b && oi < i)) { b = ob; i = oi; } // first-index tie
      }
      if (ln == 0) bestIdx[waveRow + r * 16 + hi * 8 + j] = i;
    }
  }
}

// ---------------- kernel 4: gather output, accumulate losses + counts -------
__global__ __launch_bounds__(256) void vq_finalize(const float* __restrict__ z,
                                                   const float* __restrict__ en,
                                                   const int* __restrict__ bestIdx,
                                                   float* __restrict__ zout,
                                                   int* __restrict__ counts,
                                                   float* __restrict__ accum) {
  const int wid = threadIdx.x >> 5, lane = threadIdx.x & 31;
  const int row = blockIdx.x * 8 + wid;
  f2 v = *(const f2*)(z + row * DIM + lane * 2);
  float ss = wave_sum(v.x * v.x + v.y * v.y);
  const float r = rsqrtf(ss);
  const float znx = v.x * r, zny = v.y * r;
  const int idx = bestIdx[row];
  f2 e = *(const f2*)(en + (size_t)idx * DIM + lane * 2);
  *(f2*)(zout + row * DIM + lane * 2) = e;        // STE forward value == z_vq
  const float d1x = e.x - znx, d1y = e.y - zny;   // (z_vq - z_norm)^2
  const float d2x = znx - v.x, d2y = zny - v.y;   // (z_norm - z)^2
  float m1 = wave_sum(d1x * d1x + d1y * d1y);
  float m2 = wave_sum(d2x * d2x + d2y * d2y);
  if (lane == 0) {
    atomicAdd(&counts[idx], 1);
    atomicAdd(&accum[0], m1);
    atomicAdd(&accum[1], m2);
  }
}

// ---------------- kernel 5: scalars (vq_loss, perplexity) -------------------
__global__ __launch_bounds__(256) void vq_scalars(const int* __restrict__ counts,
                                                  const float* __restrict__ accum,
                                                  float* __restrict__ out2) {
  __shared__ float red[256];
  float h = 0.f;
  for (int k = threadIdx.x; k < KCODES; k += 256) {
    const float p = (float)counts[k] * (1.0f / (float)N_ROWS);
    h -= p * logf(p + 1e-10f);
  }
  red[threadIdx.x] = h;
  __syncthreads();
  for (int s = 128; s > 0; s >>= 1) {
    if (threadIdx.x < s) red[threadIdx.x] += red[threadIdx.x + s];
    __syncthreads();
  }
  if (threadIdx.x == 0) {
    const float invND = 1.0f / ((float)N_ROWS * (float)DIM);
    out2[0] = (1.0f + BETA_F) * accum[0] * invND + BETA_F * accum[1] * invND;
    out2[1] = expf(red[0]);
  }
}

// ---------------- host-side launch ------------------------------------------
extern "C" void kernel_launch(void* const* d_in, const int* in_sizes, int n_in,
                              void* d_out, int out_size, void* d_ws, size_t ws_size,
                              hipStream_t stream) {
  (void)in_sizes; (void)n_in; (void)out_size; (void)ws_size;
  const float* z   = (const float*)d_in[0];   // [16,2048,64] fp32
  const float* emb = (const float*)d_in[1];   // [8192,64]   fp32
  float* out = (float*)d_out;                 // 32768*64 z_out, then 2 scalars

  // workspace carve-out (~7.3 MB)
  char* w = (char*)d_ws;
  bf16_t* zb = (bf16_t*)w;  w += (size_t)N_ROWS * DIM * 2;   // 4 MB
  float*  en = (float*)w;   w += (size_t)KCODES * DIM * 4;   // 2 MB
  bf16_t* eb = (bf16_t*)w;  w += (size_t)KCODES * DIM * 2;   // 1 MB
  int*    bi = (int*)w;     w += (size_t)N_ROWS * 4;         // 128 KB
  int*    counts = (int*)w; w += (size_t)KCODES * 4;         // 32 KB
  float*  accum  = (float*)w;                                //  8 B (+pad)

  // zero the accumulation state every call (counts + accum are contiguous)
  (void)hipMemsetAsync(counts, 0, (size_t)KCODES * 4 + 256, stream);

  vq_nrm_z<<<N_ROWS / 8, 256, 0, stream>>>(z, zb);
  vq_nrm_e<<<KCODES / 8, 256, 0, stream>>>(emb, en, eb);
  vq_argmax<<<N_ROWS / 128, 128, 0, stream>>>(zb, eb, bi);
  vq_finalize<<<N_ROWS / 8, 256, 0, stream>>>(z, en, bi, out, counts, accum);
  vq_scalars<<<1, 256, 0, stream>>>(counts, accum, out + (size_t)N_ROWS * DIM);
}
